// GRUForecaster_12996571037776
// MI455X (gfx1250) — matrix operand order repaired and checked by reading, not compiled, not run on hardware
//
#include <hip/hip_runtime.h>

// GRU forecaster: BATCH=64, SEQ=2048, IN=64, HIDDEN=256, HORIZON=24
// Persistent single-workgroup scan on one WGP, bf16 WMMA GEMMs.
// W_cand fragments resident in LDS (160KB); W_gate fragments streamed from L2.

#define BATCH   64
#define SEQ     2048
#define IN_DIM  64
#define HIDDEN  256
#define GATES   512     // 2*HIDDEN
#define KDIM    320     // IN_DIM + HIDDEN
#define HORIZON 24
#define KC_CNT  10      // KDIM / 32
#define NT_G    32      // GATES / 16
#define NT_C    16      // HIDDEN / 16

typedef __attribute__((ext_vector_type(16))) __bf16       v16bf;
typedef __attribute__((ext_vector_type(8)))  float        v8f;
typedef __attribute__((ext_vector_type(8)))  unsigned int v8u;

__device__ __forceinline__ unsigned short f2bf(float f) {
  unsigned int u = __float_as_uint(f);
  u += 0x7FFFu + ((u >> 16) & 1u);          // round-to-nearest-even
  return (unsigned short)(u >> 16);
}

__device__ __forceinline__ float bf2f(unsigned short s) {
  return __uint_as_float(((unsigned int)s) << 16);
}

__device__ __forceinline__ unsigned int pack2bf(float lo, float hi) {
  return (unsigned int)f2bf(lo) | ((unsigned int)f2bf(hi) << 16);
}

__device__ __forceinline__ v16bf frag_from(uint4 a, uint4 b) {
  v8u p;
  p[0] = a.x; p[1] = a.y; p[2] = a.z; p[3] = a.w;
  p[4] = b.x; p[5] = b.y; p[6] = b.z; p[7] = b.w;
  return __builtin_bit_cast(v16bf, p);
}

__device__ __forceinline__ float sigmoidf_(float x) {
  return 1.0f / (1.0f + __expf(-x));
}

// ---------------------------------------------------------------------------
// Prep: convert f32 weights -> bf16 WMMA B-fragments.
// Fragment (nt, kc) covers B[kc*32 .. kc*32+31][nt*16 .. nt*16+15].
// Per fragment: 32 lanes x 8 dwords; lane l dword v packs
//   low = B[k, n], high = B[k+1, n],
//   k = kc*32 + (l>=16 ? 16 : 0) + 2v, n = nt*16 + (l & 15).
// GEMM-side: each lane does two contiguous 16B loads per fragment.
// ---------------------------------------------------------------------------
__global__ void gru_prep_weights(const float* __restrict__ Wg,
                                 const float* __restrict__ Wc,
                                 unsigned int* __restrict__ frags) {
  int idx = blockIdx.x * blockDim.x + threadIdx.x;   // 480*256 total
  int f   = idx >> 8;
  if (f >= NT_G * KC_CNT + NT_C * KC_CNT) return;
  int rem  = idx & 255;
  int lane = rem >> 3;
  int v    = rem & 7;

  const float* W; int stride, nt, kc;
  if (f < NT_G * KC_CNT) {
    W = Wg; stride = GATES;  nt = f / KC_CNT;           kc = f % KC_CNT;
  } else {
    int g = f - NT_G * KC_CNT;
    W = Wc; stride = HIDDEN; nt = g / KC_CNT;           kc = g % KC_CNT;
  }
  int k = kc * 32 + ((lane >= 16) ? 16 : 0) + 2 * v;
  int n = nt * 16 + (lane & 15);
  frags[idx] = pack2bf(W[(size_t)k * stride + n], W[(size_t)(k + 1) * stride + n]);
}

// ---------------------------------------------------------------------------
// Persistent GRU scan: one workgroup (32 wave32 waves), whole sequence.
// ---------------------------------------------------------------------------
__global__ void __launch_bounds__(1024)
gru_scan(const float* __restrict__ x,
         const float* __restrict__ b_gate,
         const float* __restrict__ b_cand,
         const float* __restrict__ W_out,
         const float* __restrict__ b_out,
         const unsigned int* __restrict__ frags,
         float* __restrict__ out) {
  __shared__ __align__(16) unsigned int   lds_A[64 * 160];          // bf16 [64 x 320]
  __shared__ __align__(16) unsigned int   lds_wc[NT_C * KC_CNT * 256]; // Wc frags, 160KB
  __shared__ float          lds_h[64 * HIDDEN];                     // f32 h
  __shared__ unsigned short lds_z[64 * HIDDEN];                     // bf16 z
  __shared__ float          lds_bg[GATES];
  __shared__ float          lds_bc[HIDDEN];

  const int tid  = threadIdx.x;
  const int wave = tid >> 5;
  const int lane = tid & 31;
  const int lhi  = (lane >> 4) & 1;
  const int llo  = lane & 15;

  const unsigned int* wgF = frags;
  const unsigned int* wcF = frags + (size_t)NT_G * KC_CNT * 256;

  // ---- init: biases, h = 0, A h-region = 0, Wc frags -> LDS, stage x_0 ----
  if (tid < GATES)  lds_bg[tid] = b_gate[tid];
  if (tid < HIDDEN) lds_bc[tid] = b_cand[tid];
  for (int i = tid; i < 64 * HIDDEN; i += 1024) {
    lds_h[i] = 0.0f;
    lds_z[i] = 0;
  }
  for (int i = tid; i < 64 * 128; i += 1024) {
    int row = i >> 7, c = i & 127;
    lds_A[row * 160 + 32 + c] = 0u;
  }
  for (int i = tid; i < NT_C * KC_CNT * 64; i += 1024)               // 16B chunks
    ((uint4*)lds_wc)[i] = ((const uint4*)wcF)[i];
  for (int dw = tid; dw < 64 * 32; dw += 1024) {
    int row = dw >> 5, cp = dw & 31;
    const float* xp = x + ((size_t)row * SEQ + 0) * IN_DIM + cp * 2;
    lds_A[row * 160 + cp] = pack2bf(xp[0], xp[1]);
  }

  for (int t = 0; t < SEQ; ++t) {
    __syncthreads();  // (1) A-buffer [x_t, h] ready

    // Opaque 32-bit offsets: defeat LICM (weights must be re-streamed each
    // step, not hoisted into VGPRs and spilled) WITHOUT breaking clang's
    // address-space inference -> keeps global_load_b128 (LOADcnt only),
    // not flat_load_b128 (LOADcnt+DScnt).
    int wg_op = 0;
    asm volatile("" : "+v"(wg_op));
    int wc_op = 0;
    asm volatile("" : "+v"(wc_op));
    const unsigned int* wg_t = wgF + wg_op;

    // ---- gates GEMM: C[64,512] = A[64,320] @ Wg; wave owns nt = wave ----
    v8f accg[4];
    for (int i = 0; i < 4; ++i)
      for (int e = 0; e < 8; ++e) accg[i][e] = 0.0f;
    const int ntg = wave;
#pragma unroll
    for (int kc = 0; kc < KC_CNT; ++kc) {
      // batch-load all A fragments first (clause of ds_load_b128)
      uint4 a0[4], a1[4];
#pragma unroll
      for (int mt = 0; mt < 4; ++mt) {
        const uint4* ap =
            (const uint4*)&lds_A[(mt * 16 + llo) * 160 + kc * 16 + (lhi << 3)];
        a0[mt] = ap[0];
        a1[mt] = ap[1];
      }
      const uint4* bp =
          (const uint4*)(wg_t + ((size_t)(ntg * KC_CNT + kc)) * 256 + lane * 8);
      v16bf bfrag = frag_from(bp[0], bp[1]);
#pragma unroll
      for (int mt = 0; mt < 4; ++mt)
        accg[mt] = __builtin_amdgcn_wmma_f32_16x16x32_bf16(
            false, frag_from(a0[mt], a1[mt]), false, bfrag,
            (short)0, accg[mt], false, false);
    }
    __syncthreads();  // (2) all gate reads of A done

    // ---- gate epilogue: r -> bf16(r*h) into A-buffer; z -> lds_z(bf16) ----
    {
      int n = ntg * 16 + llo;                 // gate column 0..511
      if (ntg < 16) {                         // r half
#pragma unroll
        for (int mt = 0; mt < 4; ++mt)
          for (int v = 0; v < 8; ++v) {
            int m = mt * 16 + lhi * 8 + v;
            float s  = sigmoidf_(accg[mt][v] + lds_bg[n]);
            float rh = s * lds_h[m * HIDDEN + n];
            ((unsigned short*)lds_A)[m * 320 + 64 + n] = f2bf(rh);
          }
      } else {                                // z half
        int zc = n - 256;
#pragma unroll
        for (int mt = 0; mt < 4; ++mt)
          for (int v = 0; v < 8; ++v) {
            int m = mt * 16 + lhi * 8 + v;
            lds_z[m * HIDDEN + zc] = f2bf(sigmoidf_(accg[mt][v] + lds_bg[n]));
          }
      }
    }
    __syncthreads();  // (3) A-buffer now [x_t, r*h]; z ready

    // ---- candidate GEMM: C[64,256] = A2 @ Wc (Wc frags in LDS) ----
    v8f accc[2];
    for (int i = 0; i < 2; ++i)
      for (int e = 0; e < 8; ++e) accc[i][e] = 0.0f;
    const int ntc = wave & 15;
    const int mtb = (wave >> 4) * 2;
#pragma unroll
    for (int kc = 0; kc < KC_CNT; ++kc) {
      uint4 a0[2], a1[2];
#pragma unroll
      for (int i = 0; i < 2; ++i) {
        const uint4* ap = (const uint4*)
            &lds_A[((mtb + i) * 16 + llo) * 160 + kc * 16 + (lhi << 3)];
        a0[i] = ap[0];
        a1[i] = ap[1];
      }
      const uint4* bp = (const uint4*)
          &lds_wc[(ntc * KC_CNT + kc) * 256 + lane * 8 + wc_op];
      v16bf bfrag = frag_from(bp[0], bp[1]);
#pragma unroll
      for (int i = 0; i < 2; ++i)
        accc[i] = __builtin_amdgcn_wmma_f32_16x16x32_bf16(
            false, frag_from(a0[i], a1[i]), false, bfrag,
            (short)0, accc[i], false, false);
    }
    __syncthreads();  // (4) all cand reads of A done

    // ---- blend: h = (1-z)*tanh(c + bc) + z*h ----
    {
      int n = ntc * 16 + llo;                 // hidden column 0..255
#pragma unroll
      for (int i = 0; i < 2; ++i)
        for (int v = 0; v < 8; ++v) {
          int m = (mtb + i) * 16 + lhi * 8 + v;
          float ht = tanhf(accc[i][v] + lds_bc[n]);
          float z  = bf2f(lds_z[m * HIDDEN + n]);
          float h0 = lds_h[m * HIDDEN + n];
          float hn = (1.0f - z) * ht + z * h0;
          lds_h[m * HIDDEN + n] = hn;
          ((unsigned short*)lds_A)[m * 320 + 64 + n] = f2bf(hn);
        }
    }

    // ---- stage x_{t+1} into A-buffer cols 0..63 ----
    if (t + 1 < SEQ) {
      for (int dw = tid; dw < 64 * 32; dw += 1024) {
        int row = dw >> 5, cp = dw & 31;
        const float* xp = x + ((size_t)row * SEQ + (t + 1)) * IN_DIM + cp * 2;
        lds_A[row * 160 + cp] = pack2bf(xp[0], xp[1]);
      }
    }
  }
  __syncthreads();

  // ---- output projection: out[64,24] = h[64,256] @ W_out + b_out ----
  for (int o = tid; o < BATCH * HORIZON; o += 1024) {
    int b = o / HORIZON, j = o % HORIZON;
    float s = b_out[j];
    for (int k = 0; k < HIDDEN; ++k)
      s += lds_h[b * HIDDEN + k] * W_out[k * HORIZON + j];
    out[o] = s;
  }
}

// ---------------------------------------------------------------------------
extern "C" void kernel_launch(void* const* d_in, const int* in_sizes, int n_in,
                              void* d_out, int out_size, void* d_ws, size_t ws_size,
                              hipStream_t stream) {
  const float* x      = (const float*)d_in[0];
  const float* W_gate = (const float*)d_in[1];
  const float* b_gate = (const float*)d_in[2];
  const float* W_cand = (const float*)d_in[3];
  const float* b_cand = (const float*)d_in[4];
  const float* W_out  = (const float*)d_in[5];
  const float* b_out  = (const float*)d_in[6];
  unsigned int* frags = (unsigned int*)d_ws;    // 480 fragments * 1KB = 480 KB

  gru_prep_weights<<<480, 256, 0, stream>>>(W_gate, W_cand, frags);
  gru_scan<<<1, 1024, 0, stream>>>(x, b_gate, b_cand, W_out, b_out, frags,
                                   (float*)d_out);
}
